// LidarDeformableFeatureAggregation_87995289960549
// MI455X (gfx1250) — compile-verified, matrix operand order checked by reading, not run on hardware
//
#include <hip/hip_runtime.h>

// ---------------------------------------------------------------------------
// LidarDeformableFeatureAggregation for MI455X (gfx1250, wave32, WMMA)
// ---------------------------------------------------------------------------
// Shapes
static constexpr int EMBED = 256;
static constexpr int G     = 8;
static constexpr int GD    = 32;     // EMBED / G
static constexpr int NC    = 6;
static constexpr int NL    = 4;
static constexpr int NP    = 13;
static constexpr int NA    = 900;
static constexpr int NAP   = 912;    // padded to 57 * 16
static constexpr int NW    = G * NC * NL * NP;   // 2496
static constexpr int CLP   = NC * NL * NP;       // 312

// workspace layout (bytes)
static constexpr size_t OFF_FEAT   = 0;                          // bf16 (912,256)
static constexpr size_t OFF_WT     = 466944;                     // bf16 (2496,256)  w_weights^T
static constexpr size_t OFF_WOUT   = 1744896;                    // bf16 (256,256)   w_out^T
static constexpr size_t OFF_FUSED  = 1875968;                    // bf16 (912,256)
static constexpr size_t OFF_LOGITS = 2342912;                    // f32  (900,2496) logits->weights
static constexpr size_t OFF_FMT0   = 11328512;                   // bf16 (6,64*176,256)
static constexpr size_t OFF_FMT1   = 45931520;                   // bf16 (6,32*88,256)
static constexpr size_t OFF_FMT2   = 54582272;                   // bf16 (6,16*44,256)
static constexpr size_t OFF_FMT3   = 56744960;                   // bf16 (6,8*22,256)

typedef __attribute__((ext_vector_type(16))) __bf16 v16bf;
typedef __attribute__((ext_vector_type(8)))  __bf16 v8bf;
typedef __attribute__((ext_vector_type(8)))  float  v8f;

__device__ __forceinline__ int imin(int a, int b) { return a < b ? a : b; }
__device__ __forceinline__ int imax(int a, int b) { return a > b ? a : b; }

// Load one 16x32 bf16 WMMA operand fragment for this lane.
// Matrix is row-major with row stride 256 (K contiguous). `base` points at
// row start + K-chunk offset. ISA layout (16-bit A 16x32): lane-half selects
// K {0..7|8..15} in VGPR0..3 and K {16..23|24..31} in VGPR4..7.
__device__ __forceinline__ v16bf load_frag32(const __bf16* base, int half) {
  union { v16bf v; v8bf h[2]; } u;
  u.h[0] = *reinterpret_cast<const v8bf*>(base + 8 * half);
  u.h[1] = *reinterpret_cast<const v8bf*>(base + 16 + 8 * half);
  return u.v;
}

// ---------------------------------------------------------------------------
// Prep: feat = bf16(instance + anchor), zero-padded to 912 rows
__global__ __launch_bounds__(256)
void k_prep_feat(const float* __restrict__ inst, const float* __restrict__ anc,
                 __bf16* __restrict__ feat) {
  int idx = blockIdx.x * 256 + threadIdx.x;          // 912*256 threads
  feat[idx] = (idx < NA * EMBED) ? (__bf16)(inst[idx] + anc[idx]) : (__bf16)0.f;
}

// Prep: wT[n*K + k] = bf16(w[k*cols + n])  (K = 256 rows)
__global__ __launch_bounds__(256)
void k_transpose_w(const float* __restrict__ w, __bf16* __restrict__ wT, int cols) {
  int idx = blockIdx.x * 256 + threadIdx.x;
  int total = 256 * cols;
  if (idx >= total) return;
  int n = idx / 256, k = idx % 256;
  wT[idx] = (__bf16)w[(size_t)k * cols + n];
}

// Prep: feature maps (NC,256,H,W) f32 -> channels-last (NC,H*W,256) bf16
// LDS 32x32 tile transpose, coalesced both directions.
__global__ __launch_bounds__(256)
void k_fm_transpose(const float* __restrict__ src, __bf16* __restrict__ dst, int HW) {
  __shared__ float tile[32][33];
  const int c    = blockIdx.z;
  const int ch0  = blockIdx.y * 32;
  const int pix0 = blockIdx.x * 32;
  const int lx = threadIdx.x & 31;
  const int ly = threadIdx.x >> 5;     // 0..7
#pragma unroll
  for (int s = 0; s < 4; ++s) {
    int ch = ch0 + ly + s * 8;
    int pix = pix0 + lx;
    float v = 0.f;
    if (pix < HW) v = src[((size_t)c * 256 + ch) * HW + pix];
    tile[ly + s * 8][lx] = v;
  }
  __syncthreads();
#pragma unroll
  for (int s = 0; s < 4; ++s) {
    int pix = pix0 + ly + s * 8;
    int ch = ch0 + lx;
    if (pix < HW)
      dst[((size_t)c * HW + pix) * 256 + ch] = (__bf16)tile[lx][ly + s * 8];
  }
}

// ---------------------------------------------------------------------------
// GEMM1: logits(900,2496) = feat(912,256) @ w_weights(256,2496) + bias
// bf16 WMMA 16x16x32, one 16x16 tile per wave.
__global__ __launch_bounds__(256)
void k_gemm_weights(const __bf16* __restrict__ feat, const __bf16* __restrict__ wT,
                    const float* __restrict__ bias, float* __restrict__ logits) {
  const int wave = threadIdx.x >> 5;
  const int lane = threadIdx.x & 31;
  const int half = lane >> 4;
  const int l16  = lane & 15;
  const int ntile = blockIdx.y * 8 + wave;        // 0..159
  if (ntile >= NW / 16) return;                   // wave-uniform: EXEC stays all-1s
  const int a0 = blockIdx.x * 16;
  const int n0 = ntile * 16;
  const __bf16* arow = feat + (size_t)(a0 + l16) * 256;
  const __bf16* brow = wT   + (size_t)(n0 + l16) * 256;
  v8f c = {};
#pragma unroll
  for (int kc = 0; kc < 8; ++kc) {
    v16bf a = load_frag32(arow + kc * 32, half);
    v16bf b = load_frag32(brow + kc * 32, half);
    c = __builtin_amdgcn_wmma_f32_16x16x32_bf16(false, a, false, b, (short)0, c,
                                                false, false);
  }
  const int n = n0 + l16;
  const float bn = bias[n];
#pragma unroll
  for (int r = 0; r < 8; ++r) {
    int m = a0 + r + 8 * half;                    // C layout: VGPR r -> M=r / r+8
    if (m < NA) logits[(size_t)m * NW + n] = c[r] + bn;
  }
}

// ---------------------------------------------------------------------------
// Softmax over the 312 (cam,level,point) entries per (anchor, group), in place.
// One block per anchor; wave g (0..7) reduces its group with wave32 shuffles.
__global__ __launch_bounds__(256)
void k_softmax(float* __restrict__ logits) {
  __shared__ float sl[NW];
  const int a = blockIdx.x;
  for (int i = threadIdx.x; i < NW; i += 256) sl[i] = logits[(size_t)a * NW + i];
  __syncthreads();
  const int g = threadIdx.x >> 5;
  const int lane = threadIdx.x & 31;
  float mx = -3.4e38f;
  for (int clp = lane; clp < CLP; clp += 32) mx = fmaxf(mx, sl[clp * G + g]);
#pragma unroll
  for (int o = 16; o >= 1; o >>= 1) mx = fmaxf(mx, __shfl_xor(mx, o, 32));
  float sum = 0.f;
  for (int clp = lane; clp < CLP; clp += 32) sum += __expf(sl[clp * G + g] - mx);
#pragma unroll
  for (int o = 16; o >= 1; o >>= 1) sum += __shfl_xor(sum, o, 32);
  const float inv = 1.f / sum;
  for (int clp = lane; clp < CLP; clp += 32)
    logits[(size_t)a * NW + clp * G + g] = __expf(sl[clp * G + g] - mx) * inv;
}

// ---------------------------------------------------------------------------
// Project keypoints, bilinear-sample channels-last bf16 pyramid, fuse with
// softmaxed weights, sum over points. One block per anchor, thread = channel.
__global__ __launch_bounds__(256)
void k_sample_fuse(const float* __restrict__ key_points,   // (900,13,3)
                   const float* __restrict__ proj,         // (6,4,4)
                   const float* __restrict__ image_wh,     // (6,2)
                   const float* __restrict__ weights,      // (900,312,8)
                   const __bf16* __restrict__ fmT0, const __bf16* __restrict__ fmT1,
                   const __bf16* __restrict__ fmT2, const __bf16* __restrict__ fmT3,
                   __bf16* __restrict__ fused)             // (912,256)
{
  const int a = blockIdx.x;
  const int t = threadIdx.x;
  if (a >= NA) { fused[(size_t)a * 256 + t] = (__bf16)0.f; return; }

  __shared__ float swt[NW];
  __shared__ float snx[NC * NP], sny[NC * NP];
  __shared__ int   sx0[CLP], sy0[CLP];
  __shared__ float sfx[CLP], sfy[CLP];

  for (int i = t; i < NW; i += 256) swt[i] = weights[(size_t)a * NW + i];
  if (t < NC * NP) {
    int c = t / NP, p = t % NP;
    const float* kp = key_points + ((size_t)a * NP + p) * 3;
    const float x = kp[0], y = kp[1], z = kp[2];
    const float* M = proj + c * 16;
    float px = M[0] * x + M[1] * y + M[2]  * z + M[3];
    float py = M[4] * x + M[5] * y + M[6]  * z + M[7];
    float pz = M[8] * x + M[9] * y + M[10] * z + M[11];
    float d = fmaxf(pz, 1e-5f);
    snx[t] = (px / d / image_wh[c * 2 + 0]) * 2.f - 1.f;
    sny[t] = (py / d / image_wh[c * 2 + 1]) * 2.f - 1.f;
  }
  __syncthreads();

  const int LW[4] = {176, 88, 44, 22};
  const int LH[4] = {64, 32, 16, 8};
  for (int idx = t; idx < CLP; idx += 256) {
    int c = idx / (NL * NP);
    int rem = idx % (NL * NP);
    int l = rem / NP, p = rem % NP;
    float nx = snx[c * NP + p], ny = sny[c * NP + p];
    float gx = (nx + 1.f) * (LW[l] * 0.5f) - 0.5f;   // align_corners=False
    float gy = (ny + 1.f) * (LH[l] * 0.5f) - 0.5f;
    float fx0 = floorf(gx), fy0 = floorf(gy);
    sx0[idx] = (int)fx0; sy0[idx] = (int)fy0;
    sfx[idx] = gx - fx0; sfy[idx] = gy - fy0;
  }
  __syncthreads();

  const __bf16* fmt[4] = {fmT0, fmT1, fmT2, fmT3};
  const int d = t;
  const int g = d >> 5;                               // GD = 32
  float acc = 0.f;
#pragma unroll
  for (int l = 0; l < NL; ++l) {
    const int W = LW[l], H = LH[l], HW = W * H;
    const __bf16* fm = fmt[l];
    for (int c = 0; c < NC; ++c) {
      const size_t cbase = (size_t)c * HW;
      for (int p = 0; p < NP; ++p) {
        const int idx = (c * NL + l) * NP + p;
        const int x0 = sx0[idx], y0 = sy0[idx];
        const float fx = sfx[idx], fy = sfy[idx];
        const float wt = swt[idx * G + g];
        const bool xv0 = (x0 >= 0) & (x0 < W);
        const bool xv1 = (x0 + 1 >= 0) & (x0 + 1 < W);
        const bool yv0 = (y0 >= 0) & (y0 < H);
        const bool yv1 = (y0 + 1 >= 0) & (y0 + 1 < H);
        const int xc0 = imin(imax(x0, 0), W - 1);
        const int xc1 = imin(imax(x0 + 1, 0), W - 1);
        const int yc0 = imin(imax(y0, 0), H - 1);
        const int yc1 = imin(imax(y0 + 1, 0), H - 1);
        const __bf16* r0 = fm + (cbase + (size_t)yc0 * W) * 256 + d;
        const __bf16* r1 = fm + (cbase + (size_t)yc1 * W) * 256 + d;
        const float f00 = (xv0 & yv0) ? (float)r0[(size_t)xc0 * 256] : 0.f;
        const float f01 = (xv1 & yv0) ? (float)r0[(size_t)xc1 * 256] : 0.f;
        const float f10 = (xv0 & yv1) ? (float)r1[(size_t)xc0 * 256] : 0.f;
        const float f11 = (xv1 & yv1) ? (float)r1[(size_t)xc1 * 256] : 0.f;
        const float s = (f00 * (1.f - fx) + f01 * fx) * (1.f - fy) +
                        (f10 * (1.f - fx) + f11 * fx) * fy;
        acc += wt * s;
      }
    }
  }
  fused[(size_t)a * 256 + d] = (__bf16)acc;
}

// ---------------------------------------------------------------------------
// GEMM2: out(900,256) = fused(912,256) @ w_out(256,256) + b_out + instance
__global__ __launch_bounds__(256)
void k_gemm_out(const __bf16* __restrict__ fused, const __bf16* __restrict__ woutT,
                const float* __restrict__ bout, const float* __restrict__ inst,
                float* __restrict__ out) {
  const int wave = threadIdx.x >> 5;
  const int lane = threadIdx.x & 31;
  const int half = lane >> 4;
  const int l16  = lane & 15;
  const int a0 = blockIdx.x * 16;
  const int n0 = (blockIdx.y * 8 + wave) * 16;     // 16 N-tiles total
  const __bf16* arow = fused + (size_t)(a0 + l16) * 256;
  const __bf16* brow = woutT + (size_t)(n0 + l16) * 256;
  v8f c = {};
#pragma unroll
  for (int kc = 0; kc < 8; ++kc) {
    v16bf a = load_frag32(arow + kc * 32, half);
    v16bf b = load_frag32(brow + kc * 32, half);
    c = __builtin_amdgcn_wmma_f32_16x16x32_bf16(false, a, false, b, (short)0, c,
                                                false, false);
  }
  const int n = n0 + l16;
  const float bn = bout[n];
#pragma unroll
  for (int r = 0; r < 8; ++r) {
    int m = a0 + r + 8 * half;
    if (m < NA) out[(size_t)m * 256 + n] = c[r] + bn + inst[(size_t)m * 256 + n];
  }
}

// ---------------------------------------------------------------------------
extern "C" void kernel_launch(void* const* d_in, const int* in_sizes, int n_in,
                              void* d_out, int out_size, void* d_ws, size_t ws_size,
                              hipStream_t stream) {
  const float* inst = (const float*)d_in[0];
  const float* anc  = (const float*)d_in[1];
  const float* kp   = (const float*)d_in[2];
  const float* fm[4] = {(const float*)d_in[3], (const float*)d_in[4],
                        (const float*)d_in[5], (const float*)d_in[6]};
  const float* proj = (const float*)d_in[7];
  const float* iwh  = (const float*)d_in[8];
  const float* wW   = (const float*)d_in[9];
  const float* bW   = (const float*)d_in[10];
  const float* wO   = (const float*)d_in[11];
  const float* bO   = (const float*)d_in[12];
  float* out = (float*)d_out;

  char* ws = (char*)d_ws;
  __bf16* feat   = (__bf16*)(ws + OFF_FEAT);
  __bf16* wT     = (__bf16*)(ws + OFF_WT);
  __bf16* woutT  = (__bf16*)(ws + OFF_WOUT);
  __bf16* fusedb = (__bf16*)(ws + OFF_FUSED);
  float*  logits = (float*)(ws + OFF_LOGITS);
  __bf16* fmT[4] = {(__bf16*)(ws + OFF_FMT0), (__bf16*)(ws + OFF_FMT1),
                    (__bf16*)(ws + OFF_FMT2), (__bf16*)(ws + OFF_FMT3)};
  const int HWs[4] = {64 * 176, 32 * 88, 16 * 44, 8 * 22};

  // Stage 0: precision/layout prep
  k_prep_feat<<<NAP, 256, 0, stream>>>(inst, anc, feat);
  k_transpose_w<<<(256 * NW + 255) / 256, 256, 0, stream>>>(wW, wT, NW);
  k_transpose_w<<<(256 * 256 + 255) / 256, 256, 0, stream>>>(wO, woutT, 256);
  for (int l = 0; l < 4; ++l) {
    dim3 grid((HWs[l] + 31) / 32, 8, NC);
    k_fm_transpose<<<grid, 256, 0, stream>>>(fm[l], fmT[l], HWs[l]);
  }

  // Stage 1: weight GEMM (WMMA bf16) + softmax
  k_gemm_weights<<<dim3(NAP / 16, 20), 256, 0, stream>>>(feat, wT, bW, logits);
  k_softmax<<<NA, 256, 0, stream>>>(logits);

  // Stage 2: projection + bilinear sampling + weighted fusion
  k_sample_fuse<<<NAP, 256, 0, stream>>>(kp, proj, iwh, logits, fmT[0], fmT[1],
                                         fmT[2], fmT[3], fusedb);

  // Stage 3: output projection GEMM (WMMA bf16) + bias + residual
  k_gemm_out<<<dim3(NAP / 16, 2), 256, 0, stream>>>(fusedb, woutT, bO, inst, out);
}
